// ManoPoseHead_24129126269022
// MI455X (gfx1250) — compile-verified, matrix operand order checked by reading, not run on hardware
//
#include <hip/hip_runtime.h>
#include <hip/hip_bf16.h>

// ManoPoseHead fused head-MLP for gfx1250 (MI455X).
// X[16384,1408] fp32; W1[32,1408,256]; b1[32,256]; W2[32,256,3]; b2[32,3]
// out[16384, 96] fp32.
//
// bf16 WMMA (v_wmma_f32_16x16x32_bf16) for the 378-GFLOP layer-1 GEMM,
// fused ReLU + 256->3 projection (the [B,32,256] intermediate never touches
// HBM). One-time fp32->bf16 conversion puts X (46MB) and fragment-swizzled
// W1 (23MB) in workspace; both stay L2-resident (192MB). LDS double-buffer
// is filled with GLOBAL_LOAD_ASYNC_TO_LDS_B128 (ASYNCcnt-tracked), so the
// copy engine path never stalls the WMMA stream.

#define NH    32
#define DIN   1408
#define DHID  256
#define NKS   44      // 1408 / 32
#define KSTEP 32
#define BM    64      // rows per workgroup

typedef __attribute__((ext_vector_type(16))) __bf16 v16bf;
typedef __attribute__((ext_vector_type(8)))  float  v8f;

union Frag32B { uint4 q[2]; v16bf v; };

__device__ __forceinline__ unsigned int pack_bf16(float a, float b) {
  // round-to-nearest-even fp32 -> bf16, pack two into one dword (a in [15:0])
  unsigned int ua = __float_as_uint(a);
  unsigned int ub = __float_as_uint(b);
  ua = (ua + 0x7fffu + ((ua >> 16) & 1u)) >> 16;
  ub = (ub + 0x7fffu + ((ub >> 16) & 1u)) & 0xffff0000u;
  return (ua & 0xffffu) | ub;
}

__device__ __forceinline__ void async_copy_b128(unsigned lds_addr,
                                                const void* gaddr) {
  // GLOBAL_LOAD_ASYNC_TO_LDS_B128: per-lane 16B global->LDS, ASYNCcnt-tracked
  asm volatile("global_load_async_to_lds_b128 %0, %1, off"
               :
               : "v"(lds_addr), "v"(gaddr)
               : "memory");
}

__device__ __forceinline__ void wait_async0() {
  asm volatile("s_wait_asynccnt 0x0" ::: "memory");
}

// ---------------- pass 1: X fp32 -> bf16 (row-major) ----------------
__global__ __launch_bounds__(256) void mano_cvt_x_bf16(
    const float* __restrict__ in, unsigned short* __restrict__ out, long n8) {
  long i = (long)blockIdx.x * 256 + threadIdx.x;
  if (i >= n8) return;
  const float4* p = (const float4*)in + i * 2;
  float4 a = p[0];
  float4 b = p[1];
  uint4 r;
  r.x = pack_bf16(a.x, a.y);
  r.y = pack_bf16(a.z, a.w);
  r.z = pack_bf16(b.x, b.y);
  r.w = pack_bf16(b.z, b.w);
  ((uint4*)out)[i] = r;
}

// ---------------- pass 2: W1 fp32 -> bf16 WMMA-B fragment layout ----------
// Layout: Wswz[hd][ks][nt][lane][v]  (v = dword 0..7, 2 bf16 per dword)
// Fragment element i = 2v+s of lane L  <->  B[k][n] with
//   k = ks*32 + ((v>=4)?16:0) + (L>>4)*8 + (v&3)*2 + s,  n = nt*16 + (L&15)
// = the ISA 16-bit 16x32 operand layout (CDNA5 ISA).
__global__ __launch_bounds__(256) void mano_swz_w1(
    const float* __restrict__ W1, unsigned int* __restrict__ Wswz) {
  const int t    = threadIdx.x;
  const int v    = t & 7;
  const int lane = t >> 3;
  const int ks   = blockIdx.x >> 4;
  const int nt   = blockIdx.x & 15;
  const int hd   = blockIdx.y;
  const int koff = ((v >= 4) ? 16 : 0) + (lane >> 4) * 8 + (v & 3) * 2;
  const int k    = ks * KSTEP + koff;
  const int n    = nt * 16 + (lane & 15);
  const float* s = W1 + ((size_t)hd * DIN + k) * DHID + n;
  unsigned int r = pack_bf16(s[0], s[DHID]);  // k and k+1, same column n
  Wswz[((size_t)((hd * NKS + ks) * 16 + nt) * 32 + lane) * 8 + v] = r;
}

// ---------------- pass 3: fused GEMM + ReLU + 256->3 projection ----------
__global__ __launch_bounds__(256) void mano_fused_head(
    const unsigned short* __restrict__ Xb,    // [B][1408] bf16
    const unsigned int*  __restrict__ Wswz,   // [NH][44][16][32][8] dwords
    const float* __restrict__ b1,             // [NH][256]
    const float* __restrict__ W2,             // [NH][256][3]
    const float* __restrict__ b2,             // [NH][3]
    float* __restrict__ out) {                // [B][96]
  // 80B padded A rows: 16B-aligned fragments + conflict-spread banks
  __shared__ __attribute__((aligned(16))) unsigned int lds_a[2][BM * 20];
  __shared__ __attribute__((aligned(16))) unsigned int lds_b[2][4096];
  __shared__ float part[8 * 32 * 3];   // per-wave layer-2 partials

  const int tid  = threadIdx.x;
  const int lane = tid & 31;
  const int w    = tid >> 5;       // 8 waves
  const int mw   = w >> 2;         // 0..1 : 32-row group
  const int nw   = w & 3;          // 0..3 : 64-col group
  const int ll   = lane & 15;
  const int lh   = lane >> 4;
  const int head    = blockIdx.y;
  const int rowBase = blockIdx.x * BM;

  // A staging: 4 threads per row, one 16B chunk each (8 bf16 along K)
  const int arow = tid >> 2;       // 0..63
  const int aseg = tid & 3;        // 0..3
  const unsigned short* asrc =
      Xb + (size_t)(rowBase + arow) * DIN + aseg * 8;
  const unsigned int* bsrc = Wswz + (size_t)head * NKS * 4096;

  const unsigned a_lds0 =
      (unsigned)(size_t)(void*)&lds_a[0][arow * 20 + aseg * 4];
  const unsigned b_lds0 = (unsigned)(size_t)(void*)&lds_b[0][tid * 4];

  v8f acc[2][4];
  const v8f vzero = {0.f, 0.f, 0.f, 0.f, 0.f, 0.f, 0.f, 0.f};
#pragma unroll
  for (int mi = 0; mi < 2; ++mi)
#pragma unroll
    for (int ni = 0; ni < 4; ++ni) acc[mi][ni] = vzero;

  // prologue: stage k-step 0 into buffer 0 (async copies, 16B per lane)
  async_copy_b128(a_lds0, asrc);
#pragma unroll
  for (int j = 0; j < 4; ++j)
    async_copy_b128(b_lds0 + j * 4096, bsrc + (size_t)tid * 4 + j * 1024);
  wait_async0();
  __syncthreads();

  for (int ks = 0; ks < NKS; ++ks) {
    const int cur = ks & 1;
    if (ks + 1 < NKS) {
      const int nxt = cur ^ 1;
      async_copy_b128(a_lds0 + nxt * (BM * 20 * 4), asrc + (ks + 1) * KSTEP);
      const unsigned int* bg = bsrc + (size_t)(ks + 1) * 4096 + tid * 4;
      const unsigned bl = b_lds0 + nxt * 16384;
#pragma unroll
      for (int j = 0; j < 4; ++j)
        async_copy_b128(bl + j * 4096, bg + j * 1024);
      if (ks + 3 < NKS)
        __builtin_prefetch(asrc + (ks + 3) * KSTEP, 0, 3);
    }

    // WMMA fragments straight out of LDS (2x ds_load_b128 each)
    Frag32B fa[2];
#pragma unroll
    for (int mi = 0; mi < 2; ++mi) {
      int base = (mw * 32 + mi * 16 + ll) * 20 + lh * 4;
      fa[mi].q[0] = *(const uint4*)&lds_a[cur][base];
      fa[mi].q[1] = *(const uint4*)&lds_a[cur][base + 8];
    }
    Frag32B fb[4];
#pragma unroll
    for (int ni = 0; ni < 4; ++ni) {
      int idx = ((nw * 4 + ni) * 32 + lane) * 8;
      fb[ni].q[0] = *(const uint4*)&lds_b[cur][idx];
      fb[ni].q[1] = *(const uint4*)&lds_b[cur][idx + 4];
    }
#pragma unroll
    for (int mi = 0; mi < 2; ++mi)
#pragma unroll
      for (int ni = 0; ni < 4; ++ni)
        acc[mi][ni] = __builtin_amdgcn_wmma_f32_16x16x32_bf16(
            false, fa[mi].v, false, fb[ni].v, (short)0, acc[mi][ni],
            false, false);

    // my async writes into the other buffer must land before I signal
    wait_async0();
    __syncthreads();
  }

  // ---------------- epilogue: bias + ReLU + [.,256]x[256,3] ---------------
  float b1v[4];
  float w2v[4][3];
#pragma unroll
  for (int ni = 0; ni < 4; ++ni) {
    int ng = head * DHID + nw * 64 + ni * 16 + ll;   // global hidden column
    b1v[ni] = b1[ng];
    const float* wp = W2 + (size_t)ng * 3;
    w2v[ni][0] = wp[0];
    w2v[ni][1] = wp[1];
    w2v[ni][2] = wp[2];
  }

#pragma unroll
  for (int mi = 0; mi < 2; ++mi) {
#pragma unroll
    for (int e = 0; e < 8; ++e) {
      float s0 = 0.f, s1 = 0.f, s2 = 0.f;
#pragma unroll
      for (int ni = 0; ni < 4; ++ni) {
        float h = acc[mi][ni][e] + b1v[ni];
        h = fmaxf(h, 0.f);
        s0 = fmaf(h, w2v[ni][0], s0);
        s1 = fmaf(h, w2v[ni][1], s1);
        s2 = fmaf(h, w2v[ni][2], s2);
      }
      // reduce over the 16 lanes holding the hidden columns (per half)
#pragma unroll
      for (int off = 8; off >= 1; off >>= 1) {
        s0 += __shfl_xor(s0, off, 32);
        s1 += __shfl_xor(s1, off, 32);
        s2 += __shfl_xor(s2, off, 32);
      }
      int rloc = mi * 16 + lh * 8 + e;   // row within this wave's 32 rows
      if (ll < 3) {
        float v = (ll == 0) ? s0 : ((ll == 1) ? s1 : s2);
        part[(w * 32 + rloc) * 3 + ll] = v;
      }
    }
  }
  __syncthreads();

  // sum the 4 column-group waves, add b2, store
  if (tid < 192) {
    int m = tid / 3, p = tid % 3;
    int mg = m >> 5;        // which 32-row group
    int mr = m & 31;
    float s = b2[head * 3 + p];
#pragma unroll
    for (int nn = 0; nn < 4; ++nn)
      s += part[((mg * 4 + nn) * 32 + mr) * 3 + p];
    out[(size_t)(rowBase + m) * (NH * 3) + head * 3 + p] = s;
  }
}

extern "C" void kernel_launch(void* const* d_in, const int* in_sizes, int n_in,
                              void* d_out, int out_size, void* d_ws, size_t ws_size,
                              hipStream_t stream) {
  const float* X  = (const float*)d_in[0];   // [B,1408]
  const float* W1 = (const float*)d_in[1];   // [32,1408,256]
  const float* b1 = (const float*)d_in[2];   // [32,256]
  const float* W2 = (const float*)d_in[3];   // [32,256,3]
  const float* b2 = (const float*)d_in[4];   // [32,3]
  float* out = (float*)d_out;
  const int Brows = in_sizes[0] / DIN;       // 16384

  unsigned short* Xb = (unsigned short*)d_ws;                       // 46.1 MB
  unsigned int* Wswz =
      (unsigned int*)((char*)d_ws + (size_t)Brows * DIN * 2);       // 23.1 MB

  long n8 = (long)Brows * DIN / 8;
  mano_cvt_x_bf16<<<dim3((unsigned)((n8 + 255) / 256)), 256, 0, stream>>>(
      X, Xb, n8);
  mano_swz_w1<<<dim3(NKS * 16, NH), 256, 0, stream>>>(W1, Wswz);
  mano_fused_head<<<dim3(Brows / BM, NH), 256, 0, stream>>>(
      Xb, Wswz, b1, W2, b2, out);
}
